// AttentionHeads_48859547959308
// MI455X (gfx1250) — matrix operand order repaired and checked by reading, not compiled, not run on hardware
//
#include <hip/hip_runtime.h>
#include <hip/hip_bf16.h>

// ---------------------------------------------------------------------------
// Types for CDNA5 WMMA bf16 16x16x32 (wave32)
// ---------------------------------------------------------------------------
typedef __attribute__((ext_vector_type(16))) __bf16 v16bf;
typedef __attribute__((ext_vector_type(8)))  float  v8f;

#define Bsz 2
#define Tsz 2048
#define Csz 1024
#define Hsz 16
#define Dsz 64

// float -> bf16 round-to-nearest-even
__device__ __forceinline__ __bf16 f2bf(float f) {
    union { float f; unsigned u; } in; in.f = f;
    unsigned r = in.u + 0x7FFFu + ((in.u >> 16) & 1u);
    union { unsigned short s; __bf16 b; } out;
    out.s = (unsigned short)(r >> 16);
    return out.b;
}

// Async global -> LDS copy, 16 bytes per lane (ASYNCcnt-tracked).
__device__ __forceinline__ void async_copy_b128(const void* gptr, unsigned lds_off) {
    asm volatile("global_load_async_to_lds_b128 %0, %1, off"
                 :: "v"(lds_off), "v"((unsigned long long)gptr)
                 : "memory");
}
#define WAIT_ASYNC(n) asm volatile("s_wait_asynccnt " #n ::: "memory")

// LDS byte offset of a __shared__ object (LDS aperture keeps offset in addr[31:0])
__device__ __forceinline__ unsigned lds_off_of(const void* p) {
    return (unsigned)(unsigned long long)p;
}

// A fragment: 16x32 (MxK) bf16, row-major source with row stride ld.
// Contiguous k order (not the ISA table order): lane l -> m = l&15, half hs = l>>4;
// lane holds K = hs*16 + 0..15.
__device__ __forceinline__ v16bf load_frag_a(const __bf16* base, int ld, int lane) {
    int m  = lane & 15;
    int hs = lane >> 4;
    v16bf a;
#pragma unroll
    for (int e = 0; e < 16; ++e) {
        int k = hs * 16 + e;
        a[e] = base[m * ld + k];
    }
    return a;
}

// B fragment: 32x16 (KxN), memory holds B^T rows (i.e. W[n][k]), row stride ld.
// Lane l -> n = l&15, hs = l>>4; lane holds K = hs*16 + 0..15.
__device__ __forceinline__ v16bf load_frag_bt(const __bf16* base, int ld, int lane) {
    int n  = lane & 15;
    int hs = lane >> 4;
    v16bf b;
#pragma unroll
    for (int e = 0; e < 16; ++e) {
        int k = hs * 16 + e;
        b[e] = base[n * ld + k];
    }
    return b;
}

// B fragment: 32x16 (KxN), memory stored K-major (rows = K, row stride ld).
__device__ __forceinline__ v16bf load_frag_b_kmaj(const __bf16* base, int ld, int lane) {
    int n  = lane & 15;
    int hs = lane >> 4;
    v16bf b;
#pragma unroll
    for (int e = 0; e < 16; ++e) {
        int k = hs * 16 + e;
        b[e] = base[k * ld + n];
    }
    return b;
}

__device__ __forceinline__ v8f wmma_bf16(v16bf a, v16bf b, v8f c) {
    return __builtin_amdgcn_wmma_f32_16x16x32_bf16(false, a, false, b, (short)0, c,
                                                   false, false);
}

// ---------------------------------------------------------------------------
// f32 -> bf16 conversion
// ---------------------------------------------------------------------------
__global__ void cvt_f32_bf16(const float* __restrict__ src,
                             __bf16* __restrict__ dst, int n) {
    int i = blockIdx.x * blockDim.x + threadIdx.x;
    int stride = gridDim.x * blockDim.x;
    for (; i < n; i += stride) dst[i] = f2bf(src[i]);
}

// ---------------------------------------------------------------------------
// GEMM: C[M,N] = A[M,K] * B[N,K]^T   (A,B bf16, accum f32)
// Block tile 128x128, 256 threads (8 waves), wave tile 32x64, K slab 32.
// Double-buffered LDS filled with global_load_async_to_lds_b128.
// mode 0: store f32 row-major into outF
// mode 1: scatter bf16 into q/k/v laid out [B,H,T,D]
// ---------------------------------------------------------------------------
__global__ __launch_bounds__(256) void gemm_bf16(
    const __bf16* __restrict__ A, const __bf16* __restrict__ Bm,
    int M, int N, int K, int mode,
    float* __restrict__ outF,
    __bf16* __restrict__ qo, __bf16* __restrict__ ko, __bf16* __restrict__ vo) {

    __shared__ uint4 As4[2][128][4];   // 2 x (128 x 32 bf16)
    __shared__ uint4 Bs4[2][128][4];

    const int tid  = threadIdx.x;
    const int lane = tid & 31;
    const int wave = tid >> 5;
    const int wm   = wave >> 1;     // 0..3  (M direction, 32 rows each)
    const int wn   = wave & 1;      // 0..1  (N direction, 64 cols each)

    const int m0 = blockIdx.y * 128;
    const int n0 = blockIdx.x * 128;

    const unsigned ldsA = lds_off_of(&As4[0][0][0]);
    const unsigned ldsB = lds_off_of(&Bs4[0][0][0]);

    // Issue one 32-wide K slab into LDS buffer `buf` (4 async b128 per wave).
    auto issue_slab = [&](int buf, int kk) {
#pragma unroll
        for (int i = 0; i < 2; ++i) {
            int id  = tid + i * 256;
            int row = id >> 2;
            int c4  = id & 3;
            unsigned off = (unsigned)(((buf * 128 + row) * 4 + c4) * 16);
            async_copy_b128(A + (size_t)(m0 + row) * K + kk + c4 * 8, ldsA + off);
            async_copy_b128(Bm + (size_t)(n0 + row) * K + kk + c4 * 8, ldsB + off);
        }
    };

    v8f acc[2][4];
#pragma unroll
    for (int i = 0; i < 2; ++i)
#pragma unroll
        for (int j = 0; j < 4; ++j)
#pragma unroll
            for (int r = 0; r < 8; ++r) acc[i][j][r] = 0.0f;

    issue_slab(0, 0);
    int buf = 0;
    for (int kk = 0; kk < K; kk += 32) {
        if (kk + 32 < K) {
            issue_slab(buf ^ 1, kk + 32);   // prefetch next slab
            WAIT_ASYNC(0x4);                // current slab's 4 copies done
        } else {
            WAIT_ASYNC(0x0);
        }
        __syncthreads();

        const __bf16* As = reinterpret_cast<const __bf16*>(As4[buf]);
        const __bf16* Bs = reinterpret_cast<const __bf16*>(Bs4[buf]);

        v16bf afr[2], bfr[4];
#pragma unroll
        for (int i = 0; i < 2; ++i)
            afr[i] = load_frag_a(As + (wm * 32 + i * 16) * 32, 32, lane);
#pragma unroll
        for (int j = 0; j < 4; ++j)
            bfr[j] = load_frag_bt(Bs + (wn * 64 + j * 16) * 32, 32, lane);

#pragma unroll
        for (int i = 0; i < 2; ++i)
#pragma unroll
            for (int j = 0; j < 4; ++j)
                acc[i][j] = wmma_bf16(afr[i], bfr[j], acc[i][j]);

        __syncthreads();                    // done reading `buf` before overwrite
        buf ^= 1;
    }

    // Store: C fragment lane l -> col n = l&15, row = 8*(l>>4) + vgpr
    const int nn = lane & 15;
    const int hs = lane >> 4;
#pragma unroll
    for (int i = 0; i < 2; ++i) {
#pragma unroll
        for (int j = 0; j < 4; ++j) {
            int gm_base = m0 + wm * 32 + i * 16 + hs * 8;
            int gn      = n0 + wn * 64 + j * 16 + nn;
#pragma unroll
            for (int r = 0; r < 8; ++r) {
                float val = acc[i][j][r];
                int gm = gm_base + r;
                if (mode == 0) {
                    outF[(size_t)gm * N + gn] = val;
                } else {
                    int b = gm >> 11, t = gm & (Tsz - 1);
                    int which = gn >> 10;
                    int h = (gn >> 6) & (Hsz - 1);
                    int d = gn & (Dsz - 1);
                    __bf16* dst = (which == 0) ? qo : (which == 1) ? ko : vo;
                    dst[(((size_t)b * Hsz + h) * Tsz + t) * Dsz + d] = f2bf(val);
                }
            }
        }
    }
}

// ---------------------------------------------------------------------------
// Flash attention, full (non-causal).  Grid: (T/64, B*H), 128 threads (4 waves).
// Each wave owns 16 query rows; K/V streamed through double-buffered LDS via
// async-to-LDS copies.  q,k,v: [B,H,T,D] bf16.   y: [B,T,C] bf16.
// ---------------------------------------------------------------------------
__global__ __launch_bounds__(128) void attn_flash(
    const __bf16* __restrict__ q, const __bf16* __restrict__ k,
    const __bf16* __restrict__ v, __bf16* __restrict__ y) {

    __shared__ uint4  Ks4[2][64][8];       // 2 x (64 keys x 64 d bf16)
    __shared__ uint4  Vs4[2][64][8];
    __shared__ __bf16 Ps[4][16][64];       // per-wave P staging (C->A relayout)

    const int tid  = threadIdx.x;
    const int lane = tid & 31;
    const int wave = tid >> 5;
    const int nn   = lane & 15;
    const int hs   = lane >> 4;

    const int bh = blockIdx.y;             // b*H + h
    const int b  = bh >> 4;
    const int h  = bh & (Hsz - 1);
    const int q0 = blockIdx.x * 64 + wave * 16;   // first query row of this wave

    const __bf16* Qh = q + (size_t)bh * Tsz * Dsz;
    const __bf16* Kh = k + (size_t)bh * Tsz * Dsz;
    const __bf16* Vh = v + (size_t)bh * Tsz * Dsz;

    const unsigned ldsK = lds_off_of(&Ks4[0][0][0]);
    const unsigned ldsV = lds_off_of(&Vs4[0][0][0]);

    // Issue one 64-key K/V chunk into buffer `cbuf` (8 async b128 per wave).
    auto issue_chunk = [&](int cbuf, int kc) {
#pragma unroll
        for (int i = 0; i < 4; ++i) {
            int id  = tid + i * 128;
            int row = id >> 3;
            int c   = id & 7;
            unsigned off = (unsigned)(((cbuf * 64 + row) * 8 + c) * 16);
            async_copy_b128(Kh + (size_t)(kc * 64 + row) * Dsz + c * 8, ldsK + off);
            async_copy_b128(Vh + (size_t)(kc * 64 + row) * Dsz + c * 8, ldsV + off);
        }
    };

    // Q fragments (16 x 64) held in registers for the whole key sweep.
    v16bf qa[2];
#pragma unroll
    for (int s = 0; s < 2; ++s)
        qa[s] = load_frag_a(Qh + (size_t)q0 * Dsz + s * 32, Dsz, lane);

    float mrow[8], lrow[8];
    v8f ofrag[4];
#pragma unroll
    for (int r = 0; r < 8; ++r) { mrow[r] = -1e30f; lrow[r] = 0.0f; }
#pragma unroll
    for (int nt = 0; nt < 4; ++nt)
#pragma unroll
        for (int r = 0; r < 8; ++r) ofrag[nt][r] = 0.0f;

    const float scale = 0.125f;            // 1/sqrt(64)

    issue_chunk(0, 0);
    int cbuf = 0;
    for (int kc = 0; kc < Tsz / 64; ++kc) {
        if (kc + 1 < Tsz / 64) {
            issue_chunk(cbuf ^ 1, kc + 1);  // prefetch next K/V chunk
            WAIT_ASYNC(0x8);                // current chunk's 8 copies done
        } else {
            WAIT_ASYNC(0x0);
        }
        __syncthreads();

        const __bf16* Ks = reinterpret_cast<const __bf16*>(Ks4[cbuf]);
        const __bf16* Vs = reinterpret_cast<const __bf16*>(Vs4[cbuf]);

        // S = Q * K^T  -> 4 fragments of 16x16 (f32)
        v8f sfr[4];
#pragma unroll
        for (int nt = 0; nt < 4; ++nt) {
#pragma unroll
            for (int r = 0; r < 8; ++r) sfr[nt][r] = 0.0f;
#pragma unroll
            for (int s = 0; s < 2; ++s) {
                v16bf bk = load_frag_bt(Ks + (nt * 16) * Dsz + s * 32, Dsz, lane);
                sfr[nt] = wmma_bf16(qa[s], bk, sfr[nt]);
            }
#pragma unroll
            for (int r = 0; r < 8; ++r) sfr[nt][r] *= scale;
        }

        // Row max over 64 keys: partial over fragments then 16-lane xor-reduce
        float pm[8];
#pragma unroll
        for (int r = 0; r < 8; ++r) {
            pm[r] = fmaxf(fmaxf(sfr[0][r], sfr[1][r]), fmaxf(sfr[2][r], sfr[3][r]));
#pragma unroll
            for (int msk = 1; msk < 16; msk <<= 1)
                pm[r] = fmaxf(pm[r], __shfl_xor(pm[r], msk));
        }

        float mnew[8], corr[8], psum[8];
#pragma unroll
        for (int r = 0; r < 8; ++r) {
            mnew[r] = fmaxf(mrow[r], pm[r]);
            corr[r] = __expf(mrow[r] - mnew[r]);
            psum[r] = 0.0f;
        }

        // P = exp(S - mnew); accumulate row sums
#pragma unroll
        for (int nt = 0; nt < 4; ++nt)
#pragma unroll
            for (int r = 0; r < 8; ++r) {
                float p = __expf(sfr[nt][r] - mnew[r]);
                sfr[nt][r] = p;
                psum[r] += p;
            }
#pragma unroll
        for (int r = 0; r < 8; ++r) {
#pragma unroll
            for (int msk = 1; msk < 16; msk <<= 1)
                psum[r] += __shfl_xor(psum[r], msk);
            lrow[r] = lrow[r] * corr[r] + psum[r];
            mrow[r] = mnew[r];
        }

        // Rescale running O
#pragma unroll
        for (int nt = 0; nt < 4; ++nt)
#pragma unroll
            for (int r = 0; r < 8; ++r) ofrag[nt][r] *= corr[r];

        // C-layout -> A-layout for P via per-wave LDS staging
#pragma unroll
        for (int nt = 0; nt < 4; ++nt)
#pragma unroll
            for (int r = 0; r < 8; ++r)
                Ps[wave][hs * 8 + r][nt * 16 + nn] = f2bf(sfr[nt][r]);
        asm volatile("s_wait_dscnt 0x0" ::: "memory");   // wave-local DS fence

        // O += P * V
#pragma unroll
        for (int s = 0; s < 2; ++s) {
            v16bf pa = load_frag_a(&Ps[wave][0][0] + s * 32, 64, lane);
#pragma unroll
            for (int nt = 0; nt < 4; ++nt) {
                v16bf bv = load_frag_b_kmaj(Vs + (s * 32) * Dsz + nt * 16, Dsz, lane);
                ofrag[nt] = wmma_bf16(pa, bv, ofrag[nt]);
            }
        }

        __syncthreads();                    // done reading `cbuf` before overwrite
        cbuf ^= 1;
    }

    // Normalize and store y[B,T,C] (c = h*64 + d) as bf16
#pragma unroll
    for (int nt = 0; nt < 4; ++nt)
#pragma unroll
        for (int r = 0; r < 8; ++r) {
            int t = q0 + hs * 8 + r;
            int c = h * Dsz + nt * 16 + nn;
            float val = ofrag[nt][r] / lrow[r];
            y[((size_t)b * Tsz + t) * Csz + c] = f2bf(val);
        }
}

// ---------------------------------------------------------------------------
// Launcher
// ---------------------------------------------------------------------------
extern "C" void kernel_launch(void* const* d_in, const int* in_sizes, int n_in,
                              void* d_out, int out_size, void* d_ws, size_t ws_size,
                              hipStream_t stream) {
    const float* x      = (const float*)d_in[0];   // [B,T,C]
    const float* w_attn = (const float*)d_in[1];   // [3C,C]
    const float* w_out  = (const float*)d_in[2];   // [C,C]
    float* out = (float*)d_out;                    // [B,T,C] f32

    char* ws = (char*)d_ws;
    const size_t MB = 1ull << 20;
    __bf16* xb  = (__bf16*)(ws + 0 * MB);    // 8 MB  : x bf16
    __bf16* wab = (__bf16*)(ws + 8 * MB);    // 6 MB  : w_attn bf16
    __bf16* wob = (__bf16*)(ws + 14 * MB);   // 2 MB  : w_out bf16
    __bf16* qb  = (__bf16*)(ws + 16 * MB);   // 8 MB  : Q [B,H,T,D]
    __bf16* kb  = (__bf16*)(ws + 24 * MB);   // 8 MB  : K
    __bf16* vb  = (__bf16*)(ws + 32 * MB);   // 8 MB  : V
    __bf16* yb  = (__bf16*)(ws + 40 * MB);   // 8 MB  : attention out [B,T,C]

    const int nX  = Bsz * Tsz * Csz;         // 4,194,304
    const int nWA = 3 * Csz * Csz;           // 3,145,728
    const int nWO = Csz * Csz;               // 1,048,576

    cvt_f32_bf16<<<1024, 256, 0, stream>>>(x, xb, nX);
    cvt_f32_bf16<<<1024, 256, 0, stream>>>(w_attn, wab, nWA);
    cvt_f32_bf16<<<1024, 256, 0, stream>>>(w_out, wob, nWO);

    // qkv = x * w_attn^T, scattered to Q/K/V [B,H,T,D]
    gemm_bf16<<<dim3((3 * Csz) / 128, (Bsz * Tsz) / 128), 256, 0, stream>>>(
        xb, wab, Bsz * Tsz, 3 * Csz, Csz, 1, nullptr, qb, kb, vb);

    // flash attention -> y [B,T,C]
    attn_flash<<<dim3(Tsz / 64, Bsz * Hsz), 128, 0, stream>>>(qb, kb, vb, yb);

    // out = y * w_out^T  (f32 result)
    gemm_bf16<<<dim3(Csz / 128, (Bsz * Tsz) / 128), 256, 0, stream>>>(
        yb, wob, Bsz * Tsz, Csz, Csz, 0, out, nullptr, nullptr, nullptr);
}
